// GCN_84628035600551
// MI455X (gfx1250) — compile-verified, matrix operand order checked by reading, not run on hardware
//
#include <hip/hip_runtime.h>
#include <cstdint>

#define DEV __device__ __forceinline__

typedef __attribute__((ext_vector_type(16))) _Float16 v16h;
typedef __attribute__((ext_vector_type(8)))  _Float16 v8h;
typedef __attribute__((ext_vector_type(8)))  float    v8f;

static constexpr int N_PTS = 4096;
static constexpr int BATCH = 2;
static constexpr int KWIN  = 10;
static constexpr int CIN   = 256;
static constexpr int TOT   = 2048;   // 2*C*4 methods
static constexpr int NH    = 8;      // heads
static constexpr int HD    = 256;    // head dim
static constexpr int FOUT  = 512;    // fused linear output

DEV int imin(int a, int b){ return a < b ? a : b; }
DEV int imax(int a, int b){ return a > b ? a : b; }

DEV v8f vzero8f(){
  v8f r;
  #pragma unroll
  for (int i=0;i<8;++i) r[i]=0.f;
  return r;
}
DEV v16h vzero16h(){
  v16h r;
  #pragma unroll
  for (int i=0;i<16;++i) r[i]=(_Float16)0.f;
  return r;
}

DEV v8f wmma_f16(v16h a, v16h b, v8f c){
  // v_wmma_f32_16x16x32_f16  D = A*B + C  (f16 in, f32 accum)
  return __builtin_amdgcn_wmma_f32_16x16x32_f16(false, a, false, b, (short)0, c, false, false);
}

// Fragment element e -> k = (e>>3)*16 + ((lane&16)?8:0) + (e&7)
// (16-bit A 16x32 layout; B 32x16 uses the same k map with n = lane%16).
// Loads from a k-contiguous row `p` (must be 16B aligned) as two b128s.
DEV v16h frag_row(const _Float16* p, int lane){
  const int base = (lane & 16) ? 8 : 0;
  v8h lo = *(const v8h*)(p + base);
  v8h hi = *(const v8h*)(p + base + 16);
  v16h r;
  #pragma unroll
  for (int e=0;e<8;++e){ r[e]=lo[e]; r[e+8]=hi[e]; }
  return r;
}

// ---------------------------------------------------------------- SFC keys
DEV int morton3(int x, int y, int z){
  int key = 0;
  #pragma unroll
  for (int b=0;b<10;++b)
    key |= (((x>>b)&1)<<(3*b+2)) | (((y>>b)&1)<<(3*b+1)) | (((z>>b)&1)<<(3*b));
  return key;
}

DEV int hilbert3(int x0, int x1, int x2){
  int X[3] = {x0, x1, x2};
  const int M = 1 << 9;
  for (int Q = M; Q > 1; Q >>= 1){
    int P = Q - 1;
    #pragma unroll
    for (int i=0;i<3;++i){
      bool cond = (X[i] & Q) != 0;
      int t = (X[0] ^ X[i]) & P;
      int nx0 = cond ? (X[0] ^ P) : (X[0] ^ t);
      if (i != 0) X[i] = cond ? X[i] : (X[i] ^ t);
      X[0] = nx0;
    }
  }
  X[1] ^= X[0];
  X[2] ^= X[1];
  int t = 0;
  for (int Q = M; Q > 1; Q >>= 1) if (X[2] & Q) t ^= (Q - 1);
  #pragma unroll
  for (int i=0;i<3;++i) X[i] ^= t;
  int key = 0;
  for (int b=9;b>=0;--b){
    #pragma unroll
    for (int i=0;i<3;++i)
      key = (key<<1) | ((X[i]>>b)&1);
  }
  return key;
}

// One block per (method m, batch b): quantize, key, bitonic-sort (key<<13|idx)
__global__ __launch_bounds__(512) void sfc_sort_kernel(const float* __restrict__ coords,
                                                       int* __restrict__ sorted_idx){
  __shared__ unsigned long long skey[N_PTS];   // 32KB
  __shared__ float red[512];
  __shared__ float cmn[3], cscale[3];
  const int m = blockIdx.x >> 1;
  const int b = blockIdx.x & 1;
  const int t = threadIdx.x;

  for (int ax=0; ax<3; ++ax){
    const float* src = coords + (size_t)(b*3+ax)*N_PTS;
    float lmin = 3.4e38f, lmax = -3.4e38f;
    for (int i=t;i<N_PTS;i+=512){ float v=src[i]; lmin=fminf(lmin,v); lmax=fmaxf(lmax,v); }
    red[t]=lmin; __syncthreads();
    for (int s=256;s>0;s>>=1){ if(t<s) red[t]=fminf(red[t],red[t+s]); __syncthreads(); }
    if (t==0) cmn[ax]=red[0];
    __syncthreads();
    red[t]=lmax; __syncthreads();
    for (int s=256;s>0;s>>=1){ if(t<s) red[t]=fmaxf(red[t],red[t+s]); __syncthreads(); }
    if (t==0) cscale[ax]=1023.0f/(red[0]-cmn[ax]+1e-6f);
    __syncthreads();
  }

  for (int i=t;i<N_PTS;i+=512){
    int q[3];
    #pragma unroll
    for (int ax=0;ax<3;++ax){
      float v = coords[(size_t)(b*3+ax)*N_PTS + i];
      int qi = (int)floorf((v - cmn[ax]) * cscale[ax]);
      q[ax] = imin(1023, imax(0, qi));
    }
    int key;
    if      (m==0) key = morton3(q[0],q[1],q[2]);
    else if (m==1) key = morton3(q[1],q[0],q[2]);
    else if (m==2) key = hilbert3(q[0],q[1],q[2]);
    else           key = hilbert3(q[1],q[0],q[2]);
    skey[i] = (((unsigned long long)(unsigned)key) << 13) | (unsigned)i;
  }
  __syncthreads();

  for (int kk=2; kk<=N_PTS; kk<<=1){
    for (int jj=kk>>1; jj>0; jj>>=1){
      for (int i=t;i<N_PTS;i+=512){
        int ij = i ^ jj;
        if (ij > i){
          bool up = (i & kk) == 0;
          unsigned long long a = skey[i], c = skey[ij];
          if ((a > c) == up){ skey[i]=c; skey[ij]=a; }
        }
      }
      __syncthreads();
    }
  }
  for (int i=t;i<N_PTS;i+=512)
    sorted_idx[(size_t)(m*BATCH+b)*N_PTS + i] = (int)(skey[i] & 0x1FFFull);
}

// ---------------------------------------------------------------- helpers
__global__ void f32_to_f16_kernel(const float* __restrict__ s, _Float16* __restrict__ d, int n){
  int i = blockIdx.x*blockDim.x + threadIdx.x;
  if (i < n) d[i] = (_Float16)s[i];
}

// --------------------------------------------- main per-(b,n) fused layer
// build x(10x2048 f16) -> 8-head attention (WMMA) -> fused linear (WMMA)
// -> conv (WMMA) -> store h + IN statistics (atomics)
__global__ __launch_bounds__(256) void graph_attn_layer_kernel(
    const float*    __restrict__ featsIn,    // (B, 256, N)
    const int*      __restrict__ sorted_idx, // (4, B, N)
    const _Float16* __restrict__ Wlinh,      // (512, 2048)
    const float*    __restrict__ blin,       // (512)
    const _Float16* __restrict__ Wch,        // (Cout, 512)
    float*          __restrict__ hOut,       // (B, Cout, N, 10)
    float*          __restrict__ stats,      // (B*Cout, 2) sum/sumsq
    int Cout, int invpack)
{
  __shared__ __align__(16) _Float16 xbuf[KWIN * TOT];     // 40KB
  __shared__ __align__(16) _Float16 attnb[NH * 16 * 16];  // 4KB
  __shared__ __align__(16) _Float16 gbuf[16 * FOUT];      // 16KB
  __shared__ int inds[4][KWIN];

  const int tid  = threadIdx.x;
  const int lane = tid & 31;
  const int wave = tid >> 5;
  const int b = blockIdx.x >> 12;
  const int n = blockIdx.x & (N_PTS - 1);
  const int colL = lane & 15;
  const int base = (lane & 16) ? 8 : 0;

  if (tid < 4*KWIN){
    int m = tid / KWIN, j = tid % KWIN;
    int start = imax(0, n - 5);
    int end   = imin(N_PTS, n + 6);
    int pos   = imin(start + j, end - 1);
    inds[m][j] = sorted_idx[(size_t)(m*BATCH + b)*N_PTS + pos];
  }
  __syncthreads();

  { // build x: col = c2*4 + t ; t = invperm[method]; rows >=10 synthesized as 0
    const int c = tid;                       // 0..255
    const float* fb = featsIn + (size_t)(b*CIN + c)*N_PTS;
    float cent = fb[n];
    _Float16 ch = (_Float16)cent;
    #pragma unroll
    for (int j=0;j<KWIN;++j){
      int rb = j*TOT + c*4;
      xbuf[rb+0]=ch; xbuf[rb+1]=ch; xbuf[rb+2]=ch; xbuf[rb+3]=ch;
    }
    for (int m=0;m<4;++m){
      int tcol = (invpack >> (m*8)) & 0xff;
      int colb = (CIN + c)*4 + tcol;
      for (int j=0;j<KWIN;++j){
        float nv = fb[inds[m][j]];
        xbuf[j*TOT + colb] = (_Float16)(nv - cent);
      }
    }
  }
  __syncthreads();

  // ---- attention, one head per wave ----
  const int head = wave;
  const int hoff = head * HD;

  v8f sc = vzero8f();
  for (int ks=0; ks<HD; ks+=32){
    v16h fr = (colL < KWIN) ? frag_row(xbuf + colL*TOT + hoff + ks, lane) : vzero16h();
    sc = wmma_f16(fr, fr, sc);     // scores = X_h X_h^T (A and B share the row map)
  }
  #pragma unroll
  for (int v=0; v<8; ++v){
    float s = sc[v] * (1.0f/16.0f);           // /sqrt(256)
    if (colL >= KWIN) s = -1e30f;             // mask padded cols
    float mx = s;
    #pragma unroll
    for (int o=1;o<16;o<<=1) mx = fmaxf(mx, __shfl_xor(mx, o));
    float e = __expf(s - mx);
    float sum = e;
    #pragma unroll
    for (int o=1;o<16;o<<=1) sum += __shfl_xor(sum, o);
    int row = v + ((lane & 16) ? 8 : 0);
    attnb[head*256 + row*16 + colL] = (_Float16)(e / sum);
  }

  // out_h = attn @ X_h ; written back in place (disjoint cols per head)
  v16h afr;
  {
    const _Float16* ar = attnb + head*256 + colL*16;
    v8h lo = *(const v8h*)(ar + base);
    #pragma unroll
    for (int e=0;e<8;++e){ afr[e]=lo[e]; afr[e+8]=(_Float16)0.f; }  // K 16..31 = 0
  }
  for (int dt=0; dt<16; ++dt){
    int col = hoff + dt*16 + colL;
    v16h bfr;
    #pragma unroll
    for (int e=0;e<16;++e){
      int k = ((e>>3)<<4) + base + (e&7);
      bfr[e] = (k < KWIN) ? xbuf[k*TOT + col] : (_Float16)0.f;
    }
    v8f o = wmma_f16(afr, bfr, vzero8f());
    #pragma unroll
    for (int v=0;v<8;++v){
      int row = v + ((lane & 16) ? 8 : 0);
      if (row < KWIN) xbuf[row*TOT + col] = (_Float16)o[v];
    }
  }
  __syncthreads();

  // ---- fused linear: g(16x512) = xattn(16x2048) @ Wlin^T + b ----
  v8f acc[4];
  #pragma unroll
  for (int q=0;q<4;++q) acc[q] = vzero8f();
  for (int ks=0; ks<TOT; ks+=32){
    v16h a2 = (colL < KWIN) ? frag_row(xbuf + colL*TOT + ks, lane) : vzero16h();
    #pragma unroll
    for (int q=0;q<4;++q){
      int ncol = (wave*4 + q)*16 + colL;
      v16h b2 = frag_row(Wlinh + (size_t)ncol*TOT + ks, lane);
      acc[q] = wmma_f16(a2, b2, acc[q]);
    }
  }
  #pragma unroll
  for (int q=0;q<4;++q){
    int ncol = (wave*4 + q)*16 + colL;
    float bias = blin[ncol];
    #pragma unroll
    for (int v=0;v<8;++v){
      int row = v + ((lane&16)?8:0);
      gbuf[row*FOUT + ncol] = (_Float16)(acc[q][v] + bias);
    }
  }
  __syncthreads();

  // ---- conv: h = Wc(Cout x 512) @ g^T ; store + IN stats ----
  const int tpw = Cout >> 7;                 // (Cout/16)/8 waves
  for (int q=0;q<tpw;++q){
    int ot = wave*tpw + q;
    v8f hc = vzero8f();
    for (int ks=0; ks<FOUT; ks+=32){
      v16h a3 = frag_row(Wch + (size_t)(ot*16 + colL)*FOUT + ks, lane);
      v16h b3 = frag_row(gbuf + colL*FOUT + ks, lane);
      hc = wmma_f16(a3, b3, hc);
    }
    int j = colL;
    #pragma unroll
    for (int v=0;v<8;++v){
      int o = ot*16 + v + ((lane&16)?8:0);
      float val = hc[v];
      if (j < KWIN) hOut[((size_t)(b*Cout + o)*N_PTS + n)*KWIN + j] = val;
      float s  = (j<KWIN) ? val      : 0.f;
      float ss = (j<KWIN) ? val*val  : 0.f;
      #pragma unroll
      for (int o2=1;o2<16;o2<<=1){ s += __shfl_xor(s,o2); ss += __shfl_xor(ss,o2); }
      if (j == 0){
        atomicAdd(&stats[(size_t)(b*Cout + o)*2 + 0], s);
        atomicAdd(&stats[(size_t)(b*Cout + o)*2 + 1], ss);
      }
    }
  }
}

// instance-norm over (N,k) + lrelu + max over k
__global__ void normmax_kernel(const float* __restrict__ h, const float* __restrict__ stats,
                               float* __restrict__ outF, int Cout){
  int idx = blockIdx.x*blockDim.x + threadIdx.x;
  if (idx >= BATCH*Cout*N_PTS) return;
  int bo = idx / N_PTS;
  float cnt = (float)(N_PTS*KWIN);
  float mu  = stats[(size_t)bo*2+0] / cnt;
  float var = stats[(size_t)bo*2+1] / cnt - mu*mu;
  float inv = rsqrtf(var + 1e-5f);
  const float* hp = h + (size_t)idx*KWIN;
  float mx = -3.4e38f;
  #pragma unroll
  for (int j=0;j<KWIN;++j){
    float t = (hp[j]-mu)*inv;
    t = (t>0.f) ? t : 0.2f*t;
    mx = fmaxf(mx, t);
  }
  outF[idx] = mx;
}

// pack concat(feats, feats1, feats2) to (B, N, 1024) f16 (k-contiguous for WMMA B)
__global__ void f3_pack_kernel(const float* __restrict__ f0, const float* __restrict__ f1,
                               const float* __restrict__ f2, _Float16* __restrict__ f3t){
  int idx = blockIdx.x*blockDim.x + threadIdx.x;
  if (idx >= BATCH*N_PTS*1024) return;
  int c = idx & 1023;
  int n = (idx >> 10) & (N_PTS-1);
  int b = idx >> 22;
  float v;
  if      (c < 256) v = f0[(size_t)(b*256 + c      )*N_PTS + n];
  else if (c < 512) v = f1[(size_t)(b*256 + (c-256))*N_PTS + n];
  else              v = f2[(size_t)(b*512 + (c-512))*N_PTS + n];
  f3t[idx] = (_Float16)v;
}

// h3 = W_conv3(256x1024) @ f3 ; one 16x16 tile per wave
__global__ __launch_bounds__(256) void conv3_kernel(const _Float16* __restrict__ Wc3h,
                                                    const _Float16* __restrict__ f3t,
                                                    float* __restrict__ h3,
                                                    float* __restrict__ stats){
  const int lane = threadIdx.x & 31;
  const int wave = threadIdx.x >> 5;
  const int gid  = blockIdx.x*8 + wave;      // 8192 tiles: b(2) x ot(16) x nt(256)
  const int b  = gid >> 12;
  const int ot = (gid >> 8) & 15;
  const int nt = gid & 255;
  const int colL = lane & 15;
  v8f acc = vzero8f();
  const _Float16* arow = Wc3h + (size_t)(ot*16 + colL)*1024;
  const _Float16* brow = f3t  + ((size_t)b*N_PTS + nt*16 + colL)*1024;
  for (int ks=0; ks<1024; ks+=32){
    v16h a  = frag_row(arow + ks, lane);
    v16h bb = frag_row(brow + ks, lane);
    acc = wmma_f16(a, bb, acc);
  }
  int nn = nt*16 + colL;
  #pragma unroll
  for (int v=0;v<8;++v){
    int o = ot*16 + v + ((lane&16)?8:0);
    float val = acc[v];
    h3[(size_t)(b*256 + o)*N_PTS + nn] = val;
    float s = val, ss = val*val;
    #pragma unroll
    for (int o2=1;o2<16;o2<<=1){ s += __shfl_xor(s,o2); ss += __shfl_xor(ss,o2); }
    if (colL == 0){
      atomicAdd(&stats[(size_t)(b*256 + o)*2 + 0], s);
      atomicAdd(&stats[(size_t)(b*256 + o)*2 + 1], ss);
    }
  }
}

__global__ void final_norm_kernel(const float* __restrict__ h3, const float* __restrict__ stats,
                                  float* __restrict__ outp){
  int idx = blockIdx.x*blockDim.x + threadIdx.x;
  if (idx >= BATCH*256*N_PTS) return;
  int bo = idx >> 12;
  float cnt = (float)N_PTS;
  float mu  = stats[(size_t)bo*2+0] / cnt;
  float var = stats[(size_t)bo*2+1] / cnt - mu*mu;
  float inv = rsqrtf(var + 1e-5f);
  float t = (h3[idx]-mu)*inv;
  outp[idx] = (t>0.f) ? t : 0.2f*t;
}

extern "C" void kernel_launch(void* const* d_in, const int* in_sizes, int n_in,
                              void* d_out, int out_size, void* d_ws, size_t ws_size,
                              hipStream_t stream){
  const float* coords  = (const float*)d_in[0];
  const float* feats   = (const float*)d_in[1];
  const float* W_lin1  = (const float*)d_in[2];
  const float* b_lin1  = (const float*)d_in[3];
  const float* W_lin2  = (const float*)d_in[4];
  const float* b_lin2  = (const float*)d_in[5];
  const float* W_conv1 = (const float*)d_in[6];
  const float* W_conv2 = (const float*)d_in[7];
  const float* W_conv3 = (const float*)d_in[8];
  float* outp = (float*)d_out;

  char* ws = (char*)d_ws;
  size_t off = 0;
  auto take = [&](size_t bytes)->char* {
    char* p = ws + off; off = (off + bytes + 255) & ~(size_t)255; return p;
  };
  int*      sidx   = (int*)take((size_t)4*BATCH*N_PTS*sizeof(int));
  float*    statsA = (float*)take(4096*sizeof(float));
  _Float16* Wl1h   = (_Float16*)take((size_t)512*2048*2);
  _Float16* Wl2h   = (_Float16*)take((size_t)512*2048*2);
  _Float16* Wc1h   = (_Float16*)take((size_t)256*512*2);
  _Float16* Wc2h   = (_Float16*)take((size_t)512*512*2);
  _Float16* Wc3h   = (_Float16*)take((size_t)256*1024*2);
  float*    h1     = (float*)take((size_t)BATCH*256*N_PTS*KWIN*4);
  float*    h2     = (float*)take((size_t)BATCH*512*N_PTS*KWIN*4);
  float*    feats1 = (float*)take((size_t)BATCH*256*N_PTS*4);
  float*    feats2 = (float*)take((size_t)BATCH*512*N_PTS*4);
  _Float16* f3t    = (_Float16*)take((size_t)BATCH*N_PTS*1024*2);
  float*    h3     = (float*)take((size_t)BATCH*256*N_PTS*4);
  float* stats1 = statsA, *stats2 = statsA + 1024, *stats3 = statsA + 3072;

  (void)hipMemsetAsync(statsA, 0, 4096*sizeof(float), stream);

  auto cvt = [&](const float* s, _Float16* d, int ne){
    f32_to_f16_kernel<<<(ne+255)/256, 256, 0, stream>>>(s, d, ne);
  };
  cvt(W_lin1, Wl1h, 512*2048);
  cvt(W_lin2, Wl2h, 512*2048);
  cvt(W_conv1, Wc1h, 256*512);
  cvt(W_conv2, Wc2h, 512*512);
  cvt(W_conv3, Wc3h, 256*1024);

  sfc_sort_kernel<<<8, 512, 0, stream>>>(coords, sidx);

  const int INVP1 = 0x02000301;  // invperm of perm1=[2,0,3,1] -> [1,3,0,2]
  const int INVP2 = 0x01030002;  // invperm of perm2=[1,3,0,2] -> [2,0,3,1]

  graph_attn_layer_kernel<<<BATCH*N_PTS, 256, 0, stream>>>(feats,  sidx, Wl1h, b_lin1, Wc1h, h1, stats1, 256, INVP1);
  normmax_kernel<<<(BATCH*256*N_PTS+255)/256, 256, 0, stream>>>(h1, stats1, feats1, 256);
  graph_attn_layer_kernel<<<BATCH*N_PTS, 256, 0, stream>>>(feats1, sidx, Wl2h, b_lin2, Wc2h, h2, stats2, 512, INVP2);
  normmax_kernel<<<(BATCH*512*N_PTS+255)/256, 256, 0, stream>>>(h2, stats2, feats2, 512);
  f3_pack_kernel<<<(BATCH*N_PTS*1024+255)/256, 256, 0, stream>>>(feats, feats1, feats2, f3t);
  conv3_kernel<<<1024, 256, 0, stream>>>(Wc3h, f3t, h3, stats3);
  final_norm_kernel<<<(BATCH*256*N_PTS+255)/256, 256, 0, stream>>>(h3, stats3, outp);

  (void)in_sizes; (void)n_in; (void)out_size; (void)ws_size;
}
